// Find_Module_26414048871051
// MI455X (gfx1250) — compile-verified, matrix operand order checked by reading, not run on hardware
//
#include <hip/hip_runtime.h>
#include <hip/hip_bf16.h>
#include <math.h>

// ---------------- problem dims ----------------
constexpr int BN   = 128;    // batch
constexpr int LSEQ = 256;    // sequence length
constexpr int LQ   = 8;      // query length
constexpr int ED   = 300;    // E = H = D = 300
constexpr int G4   = 1200;   // 4*H gates
constexpr int KP   = 320;    // padded K for E/H/D (mult of 32)
constexpr int K2P  = 640;    // padded K for 2H / [x;h]
constexpr int NP   = 320;    // padded output cols for D   (mult of 64)
constexpr int GP   = 1216;   // padded output cols for 4H  (mult of 64)
constexpr float EPSN = 1e-12f;

typedef __attribute__((ext_vector_type(16))) __bf16 v16bf;
typedef __attribute__((ext_vector_type(8)))  float  v8f;

// ---------------- WMMA helpers ----------------
__device__ __forceinline__ v16bf load_frag(const __bf16* p) {
  // 16 bf16 = two 16-byte loads (elements [0..7] at p, [8..15] at p+16)
  union { uint4 q[2]; v16bf v; } u;
  u.q[0] = *reinterpret_cast<const uint4*>(p);
  u.q[1] = *reinterpret_cast<const uint4*>(p + 16);
  return u.v;
}

__device__ __forceinline__ v8f wmma_bf16(v16bf a, v16bf b, v8f c) {
  return __builtin_amdgcn_wmma_f32_16x16x32_bf16(false, a, false, b, (short)0, c, false, false);
}

__device__ __forceinline__ float sigm(float x) { return 1.f / (1.f + expf(-x)); }

// ---------------- generic GEMM: C[M,Npad] = A[M,Kpad](bf16) @ B[Npad,Kpad]^T + bias ----------------
// Wave computes a 16(M) x 64(N) tile with 4 accumulators; A fragment loaded once
// per K-chunk. Explicit double-buffering: next chunk's 5 fragments are issued
// into fresh registers before the current chunk's 4 WMMAs, so the compiler can
// use partial s_wait_loadcnt and overlap load latency with matrix math.
// Block = 4 waves = 64x64 C tile. grid = (M/64, Npad/64).
// B stored row-major [Npad, Kpad], zero padded. Optional bf16 mirror of C.
// Cols >= Nreal forced to 0.
__global__ void gemm_bf16(const __bf16* __restrict__ A, int sA,
                          const __bf16* __restrict__ B, int sB,
                          float* __restrict__ C, int sC,
                          __bf16* __restrict__ Cbf, int sCbf,
                          const float* __restrict__ bias,
                          int Nreal, int Kpad) {
  const int lane = threadIdx.x & 31;
  const int wave = threadIdx.x >> 5;
  const int m0 = (blockIdx.x * 4 + wave) * 16;
  const int n0 = blockIdx.y * 64;
  const int lr = lane & 15, hi = lane >> 4;
  const __bf16* aRow = A + (size_t)(m0 + lr) * sA + hi * 8;
  const __bf16* bRow = B + (size_t)(n0 + lr) * sB + hi * 8;
  v8f acc[4] = {{}, {}, {}, {}};

  v16bf aCur = load_frag(aRow);
  v16bf bCur[4];
#pragma unroll
  for (int j = 0; j < 4; ++j) bCur[j] = load_frag(bRow + (size_t)j * 16 * sB);

  for (int kc = 32; kc < Kpad; kc += 32) {
    v16bf aNxt = load_frag(aRow + kc);
    v16bf bNxt[4];
#pragma unroll
    for (int j = 0; j < 4; ++j) bNxt[j] = load_frag(bRow + (size_t)j * 16 * sB + kc);
#pragma unroll
    for (int j = 0; j < 4; ++j) acc[j] = wmma_bf16(aCur, bCur[j], acc[j]);
    aCur = aNxt;
#pragma unroll
    for (int j = 0; j < 4; ++j) bCur[j] = bNxt[j];
  }
#pragma unroll
  for (int j = 0; j < 4; ++j) acc[j] = wmma_bf16(aCur, bCur[j], acc[j]);

#pragma unroll
  for (int j = 0; j < 4; ++j) {
    const int n = n0 + j * 16 + lr;
    const float bv = (bias && n < Nreal) ? bias[n] : 0.f;
#pragma unroll
    for (int r = 0; r < 8; ++r) {
      const int mm = m0 + hi * 8 + r;
      const float v = (n < Nreal) ? (acc[j][r] + bv) : 0.f;
      C[(size_t)mm * sC + n] = v;
      if (Cbf) Cbf[(size_t)mm * sCbf + n] = (__bf16)v;
    }
  }
}

// ---------------- LSTM step GEMM: gates[128,1216] = [x_t ; h] @ Wcomb^T ----------------
// A(k<320) from x_bf[n, t, k]; A(k>=320) from h_cur[n, k-320]. Wcomb [1216, 640] bf16.
// Same 16x64-per-wave blocking + double buffering. grid = (2, 19).
__global__ void lstm_step_gemm(const __bf16* __restrict__ xb, int t, int T,
                               const __bf16* __restrict__ h,
                               const __bf16* __restrict__ W,
                               float* __restrict__ gates) {
  const int lane = threadIdx.x & 31;
  const int wave = threadIdx.x >> 5;
  const int m0 = (blockIdx.x * 4 + wave) * 16;
  const int n0 = blockIdx.y * 64;
  const int lr = lane & 15, hi = lane >> 4;
  const __bf16* xRow = xb + ((size_t)(m0 + lr) * T + t) * KP + hi * 8;
  const __bf16* hRow = h  + (size_t)(m0 + lr) * KP + hi * 8;
  const __bf16* wRow = W  + (size_t)(n0 + lr) * K2P + hi * 8;
  v8f acc[4] = {{}, {}, {}, {}};

  v16bf aCur = load_frag(xRow);
  v16bf bCur[4];
#pragma unroll
  for (int j = 0; j < 4; ++j) bCur[j] = load_frag(wRow + (size_t)j * 16 * K2P);

  for (int c = 1; c < 20; ++c) {
    const __bf16* ap = (c < 10) ? (xRow + c * 32) : (hRow + (c - 10) * 32);
    v16bf aNxt = load_frag(ap);
    v16bf bNxt[4];
#pragma unroll
    for (int j = 0; j < 4; ++j) bNxt[j] = load_frag(wRow + (size_t)j * 16 * K2P + c * 32);
#pragma unroll
    for (int j = 0; j < 4; ++j) acc[j] = wmma_bf16(aCur, bCur[j], acc[j]);
    aCur = aNxt;
#pragma unroll
    for (int j = 0; j < 4; ++j) bCur[j] = bNxt[j];
  }
#pragma unroll
  for (int j = 0; j < 4; ++j) acc[j] = wmma_bf16(aCur, bCur[j], acc[j]);

#pragma unroll
  for (int j = 0; j < 4; ++j) {
    const int n = n0 + j * 16 + lr;
#pragma unroll
    for (int r = 0; r < 8; ++r)
      gates[(size_t)(m0 + hi * 8 + r) * GP + n] = acc[j][r];
  }
}

// ---------------- LSTM cell (elementwise) ----------------
__global__ void lstm_cell(const float* __restrict__ gates,
                          const float* __restrict__ bih, const float* __restrict__ bhh,
                          float* __restrict__ c, __bf16* __restrict__ h,
                          __bf16* __restrict__ hcat, int t, int T, int coff) {
  const int idx = blockIdx.x * blockDim.x + threadIdx.x;
  if (idx >= BN * ED) return;
  const int n = idx / ED, d = idx - n * ED;
  const float* g = gates + (size_t)n * GP;
  const float gi = g[d]       + bih[d]       + bhh[d];
  const float gf = g[300 + d] + bih[300 + d] + bhh[300 + d];
  const float gg = g[600 + d] + bih[600 + d] + bhh[600 + d];
  const float go = g[900 + d] + bih[900 + d] + bhh[900 + d];
  const float cp = c[n * KP + d];
  const float cn = sigm(gf) * cp + sigm(gi) * tanhf(gg);
  const float hv = sigm(go) * tanhf(cn);
  c[n * KP + d] = cn;
  h[n * KP + d] = (__bf16)hv;
  hcat[((size_t)n * T + t) * K2P + coff + d] = (__bf16)hv;
}

// ---------------- embedding gather -> padded bf16 ----------------
__global__ void embed_k(const int* __restrict__ tok, const float* __restrict__ emb,
                        __bf16* __restrict__ dst, int rows) {
  const int idx = blockIdx.x * blockDim.x + threadIdx.x;
  if (idx >= rows * KP) return;
  const int r = idx / KP, d = idx - r * KP;
  const float v = (d < ED) ? emb[(size_t)tok[r] * ED + d] : 0.f;
  dst[idx] = (__bf16)v;
}

// ---------------- weight builders (fp32 -> padded bf16) ----------------
__global__ void build_wcomb(const float* __restrict__ Wih, const float* __restrict__ Whh,
                            __bf16* __restrict__ dst) {
  const int idx = blockIdx.x * blockDim.x + threadIdx.x;
  if (idx >= GP * K2P) return;
  const int r = idx / K2P, k = idx - r * K2P;
  float v = 0.f;
  if (r < G4) {
    if (k < ED) v = Wih[(size_t)r * ED + k];
    else if (k >= KP && k < KP + ED) v = Whh[(size_t)r * ED + (k - KP)];
  }
  dst[idx] = (__bf16)v;
}

__global__ void build_wc(const float* __restrict__ Wc, __bf16* __restrict__ dst) {
  const int idx = blockIdx.x * blockDim.x + threadIdx.x;
  if (idx >= NP * K2P) return;
  const int r = idx / K2P, k = idx - r * K2P;
  float v = 0.f;
  if (r < ED) {
    if (k < ED) v = Wc[(size_t)r * (2 * ED) + k];
    else if (k >= KP && k < KP + ED) v = Wc[(size_t)r * (2 * ED) + ED + (k - KP)];
  }
  dst[idx] = (__bf16)v;
}

__global__ void build_pad(const float* __restrict__ W, __bf16* __restrict__ dst,
                          int Rs, int Cs, int Rp, int Cp, int trans) {
  const int idx = blockIdx.x * blockDim.x + threadIdx.x;
  if (idx >= Rp * Cp) return;
  const int r = idx / Cp, k = idx - r * Cp;
  float v = 0.f;
  if (trans) { if (r < Cs && k < Rs) v = W[(size_t)k * Cs + r]; }
  else       { if (r < Rs && k < Cs) v = W[(size_t)r * Cs + k]; }
  dst[idx] = (__bf16)v;
}

// ---------------- attention score reduce: s[row] = sum tanh(T[row,d]) * av[d] ----------------
__global__ void att_reduce(const float* __restrict__ Tm, const float* __restrict__ av,
                           float* __restrict__ s) {
  const int row = blockIdx.x;
  __shared__ float red[128];
  float acc = 0.f;
  for (int d = threadIdx.x; d < ED; d += 128)
    acc += tanhf(Tm[(size_t)row * NP + d]) * av[d];
  red[threadIdx.x] = acc; __syncthreads();
  for (int off = 64; off; off >>= 1) {
    if (threadIdx.x < off) red[threadIdx.x] += red[threadIdx.x + off];
    __syncthreads();
  }
  if (threadIdx.x == 0) s[row] = red[0];
}

// ---------------- query softmax pooling -> bf16 query vector ----------------
__global__ void query_pool(const float* __restrict__ qs, const float* __restrict__ hsq,
                           __bf16* __restrict__ qv) {
  const int n = blockIdx.x;
  float m = -1e30f;
  for (int l = 0; l < LQ; ++l) m = fmaxf(m, qs[n * LQ + l]);
  float w[LQ], S = 0.f;
  for (int l = 0; l < LQ; ++l) { w[l] = expf(qs[n * LQ + l] - m); S += w[l]; }
  const float inv = 1.f / S;
  for (int d = threadIdx.x; d < KP; d += 128) {
    float v = 0.f;
    if (d < ED) {
      for (int l = 0; l < LQ; ++l) v += w[l] * hsq[(size_t)(n * LQ + l) * NP + d];
      v *= inv;
    }
    qv[n * KP + d] = (__bf16)v;
  }
}

// ---------------- l2-normalize pooled query ----------------
__global__ void normalize_q(const float* __restrict__ U, float* __restrict__ qhat) {
  const int n = blockIdx.x;
  __shared__ float red[128];
  float a = 0.f;
  for (int d = threadIdx.x; d < ED; d += 128) { float u = U[(size_t)n * NP + d]; a += u * u; }
  red[threadIdx.x] = a; __syncthreads();
  for (int off = 64; off; off >>= 1) {
    if (threadIdx.x < off) red[threadIdx.x] += red[threadIdx.x + off];
    __syncthreads();
  }
  const float inv = 1.f / fmaxf(sqrtf(red[0]), EPSN);
  for (int d = threadIdx.x; d < NP; d += 128)
    qhat[(size_t)n * NP + d] = (d < ED) ? U[(size_t)n * NP + d] * inv : 0.f;
}

// ---------------- sliding-window bigram pooling -> bf16 fwd/bwd ----------------
__global__ void sliding(const float* __restrict__ hs, const float* __restrict__ s,
                        __bf16* __restrict__ fwd, __bf16* __restrict__ bwd, int total) {
  const int idx = blockIdx.x * blockDim.x + threadIdx.x;
  if (idx >= total) return;
  const int row = idx / KP, d = idx - row * KP;
  const int t = row % LSEQ;
  float fv = 0.f, bv = 0.f;
  if (d < ED) {
    const float h0 = hs[(size_t)row * NP + d];
    if (t == LSEQ - 1) fv = h0;
    else {
      const float s0 = s[row], s1 = s[row + 1];
      const float mx = fmaxf(s0, s1);
      const float e0 = expf(s0 - mx), e1 = expf(s1 - mx);
      const float a = e0 / (e0 + e1);
      fv = a * h0 + (1.f - a) * hs[(size_t)(row + 1) * NP + d];
    }
    if (t == 0) bv = h0;
    else {
      const float sp = s[row - 1], s0 = s[row];
      const float mx = fmaxf(sp, s0);
      const float ep = expf(sp - mx), e0 = expf(s0 - mx);
      const float b = ep / (ep + e0);
      bv = b * hs[(size_t)(row - 1) * NP + d] + (1.f - b) * h0;
    }
  }
  fwd[(size_t)row * KP + d] = (__bf16)fv;
  bwd[(size_t)row * KP + d] = (__bf16)bv;
}

// ---------------- cosine + sw-weight accumulate into output ----------------
__global__ void cos_accum(const float* __restrict__ U, const float* __restrict__ qhat,
                          const float* __restrict__ sw, float* __restrict__ out, int j) {
  const int row = blockIdx.x;
  const int n = row / LSEQ;
  __shared__ float r1[128], r2[128];
  float a = 0.f, b = 0.f;
  for (int d = threadIdx.x; d < ED; d += 128) {
    const float u = U[(size_t)row * NP + d];
    a += u * qhat[(size_t)n * NP + d];
    b += u * u;
  }
  r1[threadIdx.x] = a; r2[threadIdx.x] = b; __syncthreads();
  for (int off = 64; off; off >>= 1) {
    if (threadIdx.x < off) { r1[threadIdx.x] += r1[threadIdx.x + off]; r2[threadIdx.x] += r2[threadIdx.x + off]; }
    __syncthreads();
  }
  if (threadIdx.x == 0) {
    const float cosv = r1[0] / fmaxf(sqrtf(r2[0]), EPSN);
    const float v = cosv * sw[j];
    if (j == 0) out[row] = v; else out[row] += v;
  }
}

// ================= host side =================
static inline dim3 gemm_grid(int M, int Npad) { return dim3(M / 64, Npad / 64); }

extern "C" void kernel_launch(void* const* d_in, const int* in_sizes, int n_in,
                              void* d_out, int out_size, void* d_ws, size_t ws_size,
                              hipStream_t stream) {
  const int*   seqs    = (const int*)d_in[0];
  const int*   queries = (const int*)d_in[1];
  const float* emb     = (const float*)d_in[2];
  const float* Wih_f = (const float*)d_in[3];
  const float* Whh_f = (const float*)d_in[4];
  const float* bih_f = (const float*)d_in[5];
  const float* bhh_f = (const float*)d_in[6];
  const float* Wih_b = (const float*)d_in[7];
  const float* Whh_b = (const float*)d_in[8];
  const float* bih_b = (const float*)d_in[9];
  const float* bhh_b = (const float*)d_in[10];
  const float* Wc  = (const float*)d_in[11];
  const float* bc  = (const float*)d_in[12];
  const float* Wa  = (const float*)d_in[13];
  const float* ba  = (const float*)d_in[14];
  const float* av  = (const float*)d_in[15];
  const float* Wf  = (const float*)d_in[16];
  const float* sw  = (const float*)d_in[17];
  float* out = (float*)d_out;

  // ---- bump allocator over workspace ----
  char* p = (char*)d_ws;
  auto alloc = [&](size_t bytes) -> void* {
    void* r = (void*)p;
    p += (bytes + 255) & ~(size_t)255;
    return r;
  };
  const int RS = BN * LSEQ;  // 32768 seq rows
  const int RQ = BN * LQ;    // 1024 query rows

  __bf16* wcomb_f = (__bf16*)alloc((size_t)GP * K2P * 2);
  __bf16* wcomb_b = (__bf16*)alloc((size_t)GP * K2P * 2);
  __bf16* wc_bf   = (__bf16*)alloc((size_t)NP * K2P * 2);
  __bf16* wa_bf   = (__bf16*)alloc((size_t)NP * KP * 2);
  __bf16* wft_bf  = (__bf16*)alloc((size_t)NP * KP * 2);
  __bf16* x_bf    = (__bf16*)alloc((size_t)RS * KP * 2);
  __bf16* xq_bf   = (__bf16*)alloc((size_t)RQ * KP * 2);
  __bf16* h_bf    = (__bf16*)alloc((size_t)BN * KP * 2);
  float*  c_f     = (float*)alloc((size_t)BN * KP * 4);
  float*  gates   = (float*)alloc((size_t)BN * GP * 4);
  __bf16* hcat    = (__bf16*)alloc((size_t)RS * K2P * 2);
  __bf16* hcatq   = (__bf16*)alloc((size_t)RQ * K2P * 2);
  float*  hs_f    = (float*)alloc((size_t)RS * NP * 4);
  __bf16* hs_bf   = (__bf16*)alloc((size_t)RS * KP * 2);
  float*  hsq_f   = (float*)alloc((size_t)RQ * NP * 4);
  __bf16* hsq_bf  = (__bf16*)alloc((size_t)RQ * KP * 2);
  float*  tmp     = (float*)alloc((size_t)RS * NP * 4);
  float*  s       = (float*)alloc((size_t)RS * 4);
  float*  qs      = (float*)alloc((size_t)RQ * 4);
  __bf16* qv_bf   = (__bf16*)alloc((size_t)BN * KP * 2);
  float*  qhat    = (float*)alloc((size_t)BN * NP * 4);
  __bf16* fwd_bf  = (__bf16*)alloc((size_t)RS * KP * 2);
  __bf16* bwd_bf  = (__bf16*)alloc((size_t)RS * KP * 2);
  (void)ws_size; (void)in_sizes; (void)n_in; (void)out_size;

  // ---- zero-init padded buffers (pad columns must stay 0) ----
  hipMemsetAsync(hcat,   0, (size_t)RS * K2P * 2, stream);
  hipMemsetAsync(hcatq,  0, (size_t)RQ * K2P * 2, stream);
  hipMemsetAsync(hs_bf,  0, (size_t)RS * KP * 2, stream);
  hipMemsetAsync(hsq_bf, 0, (size_t)RQ * KP * 2, stream);

  // ---- weights -> padded bf16 ----
  build_wcomb<<<(GP * K2P + 255) / 256, 256, 0, stream>>>(Wih_f, Whh_f, wcomb_f);
  build_wcomb<<<(GP * K2P + 255) / 256, 256, 0, stream>>>(Wih_b, Whh_b, wcomb_b);
  build_wc   <<<(NP * K2P + 255) / 256, 256, 0, stream>>>(Wc, wc_bf);
  build_pad  <<<(NP * KP + 255) / 256, 256, 0, stream>>>(Wa, wa_bf, ED, ED, NP, KP, 0);
  build_pad  <<<(NP * KP + 255) / 256, 256, 0, stream>>>(Wf, wft_bf, ED, ED, NP, KP, 1);

  // ---- embeddings ----
  embed_k<<<(RS * KP + 255) / 256, 256, 0, stream>>>(seqs, emb, x_bf, RS);
  embed_k<<<(RQ * KP + 255) / 256, 256, 0, stream>>>(queries, emb, xq_bf, RQ);

  const dim3 stepGrid(BN / 64, GP / 64);  // (2, 19)
  auto run_encoder = [&](const __bf16* xb, __bf16* hc, int T) {
    // forward direction
    hipMemsetAsync(h_bf, 0, (size_t)BN * KP * 2, stream);
    hipMemsetAsync(c_f,  0, (size_t)BN * KP * 4, stream);
    for (int t = 0; t < T; ++t) {
      lstm_step_gemm<<<stepGrid, 128, 0, stream>>>(xb, t, T, h_bf, wcomb_f, gates);
      lstm_cell<<<(BN * ED + 255) / 256, 256, 0, stream>>>(gates, bih_f, bhh_f, c_f, h_bf, hc, t, T, 0);
    }
    // backward direction
    hipMemsetAsync(h_bf, 0, (size_t)BN * KP * 2, stream);
    hipMemsetAsync(c_f,  0, (size_t)BN * KP * 4, stream);
    for (int tt = 0; tt < T; ++tt) {
      const int t = T - 1 - tt;
      lstm_step_gemm<<<stepGrid, 128, 0, stream>>>(xb, t, T, h_bf, wcomb_b, gates);
      lstm_cell<<<(BN * ED + 255) / 256, 256, 0, stream>>>(gates, bih_b, bhh_b, c_f, h_bf, hc, t, T, KP);
    }
  };

  // ---- encode seqs ----
  run_encoder(x_bf, hcat, LSEQ);
  gemm_bf16<<<gemm_grid(RS, NP), 128, 0, stream>>>(hcat, K2P, wc_bf, K2P, hs_f, NP, hs_bf, KP, bc, ED, K2P);
  gemm_bf16<<<gemm_grid(RS, NP), 128, 0, stream>>>(hs_bf, KP, wa_bf, KP, tmp, NP, (__bf16*)nullptr, 0, ba, ED, KP);
  att_reduce<<<RS, 128, 0, stream>>>(tmp, av, s);

  // ---- encode queries + pooled normalized query vector ----
  run_encoder(xq_bf, hcatq, LQ);
  gemm_bf16<<<gemm_grid(RQ, NP), 128, 0, stream>>>(hcatq, K2P, wc_bf, K2P, hsq_f, NP, hsq_bf, KP, bc, ED, K2P);
  gemm_bf16<<<gemm_grid(RQ, NP), 128, 0, stream>>>(hsq_bf, KP, wa_bf, KP, tmp, NP, (__bf16*)nullptr, 0, ba, ED, KP);
  att_reduce<<<RQ, 128, 0, stream>>>(tmp, av, qs);
  query_pool<<<BN, 128, 0, stream>>>(qs, hsq_f, qv_bf);
  gemm_bf16<<<gemm_grid(BN, NP), 128, 0, stream>>>(qv_bf, KP, wft_bf, KP, tmp, NP, (__bf16*)nullptr, 0, (const float*)nullptr, ED, KP);
  normalize_q<<<BN, 128, 0, stream>>>(tmp, qhat);

  // ---- sliding-window pooling ----
  sliding<<<(RS * KP + 255) / 256, 256, 0, stream>>>(hs_f, s, fwd_bf, bwd_bf, RS * KP);

  // ---- cosine scores, combined with sw_weight ----
  const __bf16* feats[3] = { hs_bf, fwd_bf, bwd_bf };
  for (int j = 0; j < 3; ++j) {
    gemm_bf16<<<gemm_grid(RS, NP), 128, 0, stream>>>(feats[j], KP, wft_bf, KP, tmp, NP, (__bf16*)nullptr, 0, (const float*)nullptr, ED, KP);
    cos_accum<<<RS, 128, 0, stream>>>(tmp, qhat, sw, out, j);
  }
}
